// MoleculeHead_61821759259419
// MI455X (gfx1250) — matrix operand order repaired and checked by reading, not compiled, not run on hardware
//
#include <hip/hip_runtime.h>
#include <hip/hip_bf16.h>

// CDNA5 / gfx1250 WMMA types (wave32)
typedef __attribute__((ext_vector_type(16))) _Float16 v16h;
typedef __attribute__((ext_vector_type(8)))  float    v8f;

#define Q_DIM   64
#define DIM_IN  512
#define DIM_HID 1024
#define DIM_OUT 2048
#define N_DB    65536
#define TOPK    5

// LDS tile stride for the transposed dot tile: 128 rows + 4 pad floats.
// Keeps each query row 16B-aligned for b128 async stores; pad shifts banks by 4.
#define TILE_LD 132

// ---------------------------------------------------------------------------
// Generic MLP layer: C[64,N] = act(A[64,K] @ W[K,N] + b)
// mode 0: ReLU -> f32 out.  mode 1: write Bt = 1 - 2*sigmoid(x) as f16.
// Block = 128 threads (4 waves); wave w owns M-tile w (rows 16w..16w+15).
// Grid.x = N/16.
// ---------------------------------------------------------------------------
__global__ void mlp_layer_kernel(const float* __restrict__ A,
                                 const float* __restrict__ W,
                                 const float* __restrict__ bias,
                                 float* __restrict__ outF,
                                 _Float16* __restrict__ outH,
                                 int K, int N, int mode) {
    const int lane = threadIdx.x & 31;
    const int wave = threadIdx.x >> 5;            // 0..3 -> M tile
    const int m0   = wave * 16;
    const int n0   = blockIdx.x * 16;
    const int l15  = lane & 15;
    const int hi   = (lane >= 16) ? 1 : 0;
    const int row  = m0 + l15;                    // A row (M) for this lane
    const int col  = n0 + l15;                    // C/B column (N)
    const int ka   = hi * 8;                      // A K-offset per lane half
    const int kb   = hi * 8;                      // B K-offset per lane half

    v8f acc = {};
    for (int k0 = 0; k0 < K; k0 += 32) {
        const float* arow = A + (size_t)row * K + k0 + ka;
        v16h af, bf;
#pragma unroll
        for (int e = 0; e < 8; ++e) {
            af[e]     = (_Float16)arow[e];        // K = k0+ka+e
            af[e + 8] = (_Float16)arow[16 + e];   // K = k0+ka+16+e
        }
#pragma unroll
        for (int e = 0; e < 16; ++e)
            bf[e] = (_Float16)W[(size_t)(k0 + kb + (e & 7) + ((e >> 3) << 4)) * N + col];
        acc = __builtin_amdgcn_wmma_f32_16x16x32_f16(
                  false, af, false, bf, (short)0, acc, false, false);
    }

    const float b = bias[col];
#pragma unroll
    for (int r = 0; r < 8; ++r) {
        const int orow = m0 + r + hi * 8;
        float v = acc[r] + b;
        if (mode == 0) {
            v = v > 0.0f ? v : 0.0f;
            outF[(size_t)orow * N + col] = v;
        } else {
            // Bt = 1 - 2*sigmoid(v) = (1 - e^v) / (1 + e^v); fast rcp path.
            float x = v > 30.0f ? 30.0f : (v < -30.0f ? -30.0f : v);
            float t = __expf(x);
            float bt = (1.0f - t) * __builtin_amdgcn_rcpf(1.0f + t);
            outH[(size_t)orow * N + col] = (_Float16)bt;
        }
    }
}

// ---------------------------------------------------------------------------
// L1 distance as GEMM:  dot[n,j] = mol_fps[n,:] (f32, binary) . Bt[j,:] (f16)
// Bt[j][k] = 1 - 2*fp[j][k].  Full distance = S_q[j] + dot (S_q added later;
// constant per query -> does not affect top-k ranking).
// Block = 256 threads (8 waves); wave owns 16 db rows x all 64 queries.
// Result tile is transposed into LDS [query][128 rows], then drained to
// dotT[j*65536 + n] with GLOBAL_STORE_ASYNC_FROM_LDS_B128 (coalesced, DMA).
// Grid.x = 65536 / 128 = 512.
// ---------------------------------------------------------------------------
__global__ void l1dot_kernel(const float* __restrict__ db,
                             const _Float16* __restrict__ Bt,
                             float* __restrict__ dotT) {
    __shared__ float tile[Q_DIM * TILE_LD];       // 64 queries x (128+4) rows

    const int tid  = threadIdx.x;
    const int lane = tid & 31;
    const int wave = tid >> 5;
    const int l15  = lane & 15;
    const int hi   = (lane >= 16) ? 1 : 0;
    const size_t blockRow0 = (size_t)blockIdx.x * 128;
    const size_t rowbase   = blockRow0 + (size_t)wave * 16;
    const int ka = hi * 16;
    const int kb = hi * 16;

    const float*    arow = db + (rowbase + l15) * DIM_OUT + ka;
    const _Float16* brow = Bt + (size_t)l15 * DIM_OUT + kb;   // query = l15 (+16*nt)

    v8f acc0 = {}, acc1 = {}, acc2 = {}, acc3 = {};

    for (int k0 = 0; k0 < DIM_OUT; k0 += 32) {
        __builtin_prefetch(arow + k0 + 256, 0, 0);            // global_prefetch_b8
        // A fragment: two aligned 32B f32 vector loads, convert to f16
        v8f lo  = *(const v8f*)(arow + k0);
        v8f hic = *(const v8f*)(arow + k0 + 8);
        v16h af;
#pragma unroll
        for (int e = 0; e < 8; ++e) {
            af[e]     = (_Float16)lo[e];
            af[e + 8] = (_Float16)hic[e];
        }
        // B fragments: contiguous 16-half (32B) loads, one per query tile
        v16h bf0 = *(const v16h*)(brow + k0);
        v16h bf1 = *(const v16h*)(brow + (size_t)16 * DIM_OUT + k0);
        v16h bf2 = *(const v16h*)(brow + (size_t)32 * DIM_OUT + k0);
        v16h bf3 = *(const v16h*)(brow + (size_t)48 * DIM_OUT + k0);

        acc0 = __builtin_amdgcn_wmma_f32_16x16x32_f16(false, af, false, bf0, (short)0, acc0, false, false);
        acc1 = __builtin_amdgcn_wmma_f32_16x16x32_f16(false, af, false, bf1, (short)0, acc1, false, false);
        acc2 = __builtin_amdgcn_wmma_f32_16x16x32_f16(false, af, false, bf2, (short)0, acc2, false, false);
        acc3 = __builtin_amdgcn_wmma_f32_16x16x32_f16(false, af, false, bf3, (short)0, acc3, false, false);
    }

    // Transpose into LDS: tile[query][rowInBlock]
    // C layout: VGPR r -> row = r + 8*hi ; column = l15 (+16 per tile)
    const int rowInBlk = wave * 16 + hi * 8;
#pragma unroll
    for (int r = 0; r < 8; ++r) {
        tile[(l15 +  0) * TILE_LD + rowInBlk + r] = acc0[r];
        tile[(l15 + 16) * TILE_LD + rowInBlk + r] = acc1[r];
        tile[(l15 + 32) * TILE_LD + rowInBlk + r] = acc2[r];
        tile[(l15 + 48) * TILE_LD + rowInBlk + r] = acc3[r];
    }
    __syncthreads();

    // Drain LDS -> global with async DMA stores (ASYNCcnt-tracked, b128 each).
    // 64 queries * 32 b128-segments = 2048 transfers; 8 per thread.
    // Consecutive lanes hit consecutive 16B segments -> contiguous 512B runs.
#pragma unroll
    for (int k = 0; k < 8; ++k) {
        const int i   = tid + 256 * k;
        const int j   = i >> 5;                   // query
        const int seg = i & 31;                   // 16B segment within 128 rows
        const unsigned lds_off =
            (unsigned)(uintptr_t)&tile[j * TILE_LD + seg * 4];
        const unsigned long long gaddr =
            (unsigned long long)(uintptr_t)(dotT + (size_t)j * N_DB + blockRow0 + seg * 4);
        asm volatile("global_store_async_from_lds_b128 %0, %1, off"
                     :: "v"(gaddr), "v"(lds_off) : "memory");
    }
    asm volatile("s_wait_asynccnt 0x0" ::: "memory");
}

// ---------------------------------------------------------------------------
// Top-5 smallest per query. One block (256 threads) per query column.
// Reconstructs S_q[j] = (2048 - sum_k Bt[j][k]) / 2 and adds it to the 5
// emitted distances. Indices emitted as floats (d_out is float-typed).
// d_out: dist[5][64] then indices[5][64].
// ---------------------------------------------------------------------------
__global__ void topk_kernel(const float* __restrict__ dotT,
                            const _Float16* __restrict__ Bt,
                            float* __restrict__ out) {
    const int j   = blockIdx.x;
    const int tid = threadIdx.x;

    __shared__ float svals[256 * TOPK];
    __shared__ int   sidx [256 * TOPK];
    __shared__ float ssum [256];

    // partial sum of Bt row j (for S_q)
    float ps = 0.0f;
    for (int k = tid; k < DIM_OUT; k += 256)
        ps += (float)Bt[(size_t)j * DIM_OUT + k];
    ssum[tid] = ps;

    // per-thread sorted (ascending) top-5
    float bv[TOPK];
    int   bi[TOPK];
#pragma unroll
    for (int t = 0; t < TOPK; ++t) { bv[t] = 3.4e38f; bi[t] = -1; }

    const float* col = dotT + (size_t)j * N_DB;
    for (int i = tid; i < N_DB; i += 256) {
        const float v = col[i];
        if (v < bv[TOPK - 1]) {
            int p = TOPK - 1;
            while (p > 0 && v < bv[p - 1]) { bv[p] = bv[p - 1]; bi[p] = bi[p - 1]; --p; }
            bv[p] = v; bi[p] = i;
        }
    }
#pragma unroll
    for (int t = 0; t < TOPK; ++t) { svals[tid * TOPK + t] = bv[t]; sidx[tid * TOPK + t] = bi[t]; }
    __syncthreads();

    // stage 1: threads 0..7 each merge 32 lists of 5
    float mv[TOPK]; int mi[TOPK];
    if (tid < 8) {
#pragma unroll
        for (int t = 0; t < TOPK; ++t) { mv[t] = 3.4e38f; mi[t] = -1; }
        for (int s = tid * 32; s < tid * 32 + 32; ++s) {
            for (int t = 0; t < TOPK; ++t) {
                const float v = svals[s * TOPK + t];
                const int   ix = sidx[s * TOPK + t];
                if (v < mv[TOPK - 1]) {
                    int p = TOPK - 1;
                    while (p > 0 && v < mv[p - 1]) { mv[p] = mv[p - 1]; mi[p] = mi[p - 1]; --p; }
                    mv[p] = v; mi[p] = ix;
                }
            }
        }
    }
    __syncthreads();
    if (tid < 8) {
#pragma unroll
        for (int t = 0; t < TOPK; ++t) { svals[tid * TOPK + t] = mv[t]; sidx[tid * TOPK + t] = mi[t]; }
    }
    __syncthreads();

    // stage 2: thread 0 merges 8 lists of 5, adds S_q, writes outputs
    if (tid == 0) {
        float fv[TOPK]; int fi[TOPK];
#pragma unroll
        for (int t = 0; t < TOPK; ++t) { fv[t] = 3.4e38f; fi[t] = -1; }
        for (int s = 0; s < 8; ++s) {
            for (int t = 0; t < TOPK; ++t) {
                const float v = svals[s * TOPK + t];
                const int   ix = sidx[s * TOPK + t];
                if (v < fv[TOPK - 1]) {
                    int p = TOPK - 1;
                    while (p > 0 && v < fv[p - 1]) { fv[p] = fv[p - 1]; fi[p] = fi[p - 1]; --p; }
                    fv[p] = v; fi[p] = ix;
                }
            }
        }
        float sumBt = 0.0f;
        for (int s = 0; s < 256; ++s) sumBt += ssum[s];
        const float Sq = 0.5f * ((float)DIM_OUT - sumBt);
#pragma unroll
        for (int t = 0; t < TOPK; ++t) {
            out[t * Q_DIM + j]                = fv[t] + Sq;   // dist [5,64]
            out[TOPK * Q_DIM + t * Q_DIM + j] = (float)fi[t]; // indices [5,64]
        }
    }
}

// ---------------------------------------------------------------------------
// Workspace layout (bytes), all 256KB-aligned chunks:
//   X1   @ 0        : 64*1024 f32   (256 KB)
//   X2   @ 256K     : 64*1024 f32
//   X3   @ 512K     : 64*1024 f32
//   Bt   @ 768K     : 64*2048 f16   (256 KB)
//   dotT @ 1M       : 65536*64 f32  (16 MB)
// Total ~17 MB.
// ---------------------------------------------------------------------------
extern "C" void kernel_launch(void* const* d_in, const int* in_sizes, int n_in,
                              void* d_out, int out_size, void* d_ws, size_t ws_size,
                              hipStream_t stream) {
    const float* h   = (const float*)d_in[0];
    const float* mol = (const float*)d_in[1];
    const float* W1  = (const float*)d_in[2];
    const float* b1  = (const float*)d_in[3];
    const float* W2  = (const float*)d_in[4];
    const float* b2  = (const float*)d_in[5];
    const float* W3  = (const float*)d_in[6];
    const float* b3  = (const float*)d_in[7];
    const float* W4  = (const float*)d_in[8];
    const float* b4  = (const float*)d_in[9];

    char* ws = (char*)d_ws;
    float*    X1   = (float*)(ws + 0);
    float*    X2   = (float*)(ws + (256u << 10));
    float*    X3   = (float*)(ws + (512u << 10));
    _Float16* Bt   = (_Float16*)(ws + (768u << 10));
    float*    dotT = (float*)(ws + (1024u << 10));

    // MLP (WMMA f16 fragments, f32 accumulate)
    mlp_layer_kernel<<<DIM_HID / 16, 128, 0, stream>>>(h,  W1, b1, X1, nullptr, DIM_IN,  DIM_HID, 0);
    mlp_layer_kernel<<<DIM_HID / 16, 128, 0, stream>>>(X1, W2, b2, X2, nullptr, DIM_HID, DIM_HID, 0);
    mlp_layer_kernel<<<DIM_HID / 16, 128, 0, stream>>>(X2, W3, b3, X3, nullptr, DIM_HID, DIM_HID, 0);
    mlp_layer_kernel<<<DIM_OUT / 16, 128, 0, stream>>>(X3, W4, b4, nullptr, Bt, DIM_HID, DIM_OUT, 1);

    // L1 cdist as WMMA GEMM (512 MB streamed once from HBM -> bandwidth bound),
    // results transposed in LDS and drained via async LDS->global DMA stores.
    l1dot_kernel<<<N_DB / 128, 256, 0, stream>>>(mol, Bt, dotT);

    // top-5 per query + S_q correction
    topk_kernel<<<Q_DIM, 256, 0, stream>>>(dotT, Bt, (float*)d_out);
}
